// TextureEditableNeuMesh_43447889166609
// MI455X (gfx1250) — matrix operand order repaired and checked by reading, not compiled, not run on hardware
//
#include <hip/hip_runtime.h>

typedef _Float16 f16;
typedef __attribute__((ext_vector_type(16))) _Float16 v16h;
typedef __attribute__((ext_vector_type(8)))  _Float16 h8;
typedef __attribute__((ext_vector_type(8)))  float    v8f;
typedef __attribute__((ext_vector_type(4)))  float    f4;

#define KN    32      // neighbors per point
#define DF    32      // feature dim
#define HID   256     // hidden width
#define DIN   39      // D + 7 input width
#define KPAD  72      // padded f16 row: 144 B (16B aligned, bank-friendly)
#define PTS   128     // points per block == blockDim.x (4 waves)
#define NMLP  3
#define BETA  10.0f
#define EPS_R 1e-8f

union HV { h8 h[2]; v16h v; };

__device__ __forceinline__ float sigmoidf(float x) { return 1.0f / (1.0f + __expf(-x)); }

__global__ __launch_bounds__(PTS) void neumesh_fused(
    const float* __restrict__ xyz,   const float* __restrict__ vdirs,
    const float* __restrict__ mverts,const float* __restrict__ geo,
    const float* __restrict__ cfeat, const float* __restrict__ ecf,
    const float* __restrict__ rot,
    const float* __restrict__ W1,  const float* __restrict__ b1,
    const float* __restrict__ W2,  const float* __restrict__ b2,
    const float* __restrict__ sW1, const float* __restrict__ sb1,
    const float* __restrict__ sW2, const float* __restrict__ sb2,
    const int*   __restrict__ indices, const int* __restrict__ masks,
    float* __restrict__ out_sdf, float* __restrict__ out_rgb, int V)
{
  __shared__ f16   Atile[PTS][KPAD];   // per-point MLP input rows, K-padded to 64
  __shared__ f16   WT[HID][KPAD];      // W1 transposed (n-major, k-minor), zero-padded
  __shared__ float b1s[HID];
  __shared__ float w2s[HID * 3];
  __shared__ float b2s[4];
  __shared__ float rgbbuf[PTS][3];
  __shared__ int   paintf[2][PTS];
  __shared__ float scal[PTS][12];      // compact per-point scalar spill block

  const int tid  = threadIdx.x;
  const int lane = tid & 31;
  const int wave = tid >> 5;
  const int p    = blockIdx.x * PTS + tid;
  const int* __restrict__ irow = indices + (long)p * KN;   // one 128B line, L2-hot

  // ---------------- Phase 1: per-point geometry + softmax weights ----------
  const float px = xyz[p*3+0],  py = xyz[p*3+1],  pz = xyz[p*3+2];
  const float vx = vdirs[p*3+0], vy = vdirs[p*3+1], vz = vdirs[p*3+2];

  float ev[KN];          // softmax numerators (only persistent per-k array)
  {
    float dmin = 1e30f;
    float dsv[KN];       // transient: dies at end of Phase 1
    #pragma unroll
    for (int k = 0; k < KN; ++k) {
      const int v = irow[k];
      const float dx = px - mverts[v*3+0];
      const float dy = py - mverts[v*3+1];
      const float dz = pz - mverts[v*3+2];
      const float d  = sqrtf(dx*dx + dy*dy + dz*dz);
      dsv[k] = d;
      dmin = fminf(dmin, d);
    }

    float esum=0.f, sdfa=0.f, wxa=0.f, wya=0.f, wza=0.f, dma=0.f;
    unsigned mb0=0u, mb1=0u;
    float ms0=0.f, ms1=0.f, md0=0.f, md1=0.f;
    int   mc0=0, mc1=0;
    #pragma unroll
    for (int k = 0; k < KN; ++k) {
      const int v = irow[k];                              // reload (cacheline-hot)
      const float e = __expf(-BETA * (dsv[k] - dmin));    // stable softmax numerator
      ev[k] = e;
      esum += e;
      sdfa += e * geo[v];
      wxa  += e * mverts[v*3+0];
      wya  += e * mverts[v*3+1];
      wza  += e * mverts[v*3+2];
      dma  += e * dsv[k];
      if (masks[v]     > 0) { mb0 |= (1u<<k); ms0 += e; md0 += e*dsv[k]; ++mc0; }
      if (masks[V + v] > 0) { mb1 |= (1u<<k); ms1 += e; md1 += e*dsv[k]; ++mc1; }
    }
    const float inv = 1.0f / esum;
    out_sdf[p] = sdfa * inv;
    // nabla = normalize(sum_k w_k (x - v_k)) = normalize(x - sum_k w_k v_k)
    float nx = px - wxa*inv, ny = py - wya*inv, nz = pz - wza*inv;
    const float nn = 1.0f / (sqrtf(nx*nx + ny*ny + nz*nz) + EPS_R);
    nx *= nn; ny *= nn; nz *= nn;

    paintf[0][tid] = (mc0 >= 8);
    paintf[1][tid] = (mc1 >= 8);

    // stash per-point scalars; reading them back below keeps the big
    // Phase-1 transient state out of later live ranges (VGPR pressure)
    scal[tid][0]=nx; scal[tid][1]=ny; scal[tid][2]=nz;
    scal[tid][3]=inv;                      // 1/esum
    scal[tid][4]=dma*inv;                  // ds_mean
    scal[tid][5]=1.0f/(ms0 + EPS_R*esum);  // slave0 renorm
    scal[tid][6]=1.0f/(ms1 + EPS_R*esum);  // slave1 renorm
    scal[tid][7]=md0; scal[tid][8]=md1;
    scal[tid][9]=__int_as_float((int)mb0);
    scal[tid][10]=__int_as_float((int)mb1);
    scal[tid][11]=0.f;
  }

  const float nx = scal[tid][0], ny = scal[tid][1], nz = scal[tid][2];
  const float inv = scal[tid][3];
  const float dsmean = scal[tid][4];
  const float rcp0 = scal[tid][5], rcp1 = scal[tid][6];
  const float md0 = scal[tid][7], md1 = scal[tid][8];
  const unsigned mb0 = (unsigned)__float_as_int(scal[tid][9]);
  const unsigned mb1 = (unsigned)__float_as_int(scal[tid][10]);

  // ---------------- Fused MLP loop: main head + 2 slave heads --------------
  for (int mlp = 0; mlp < NMLP; ++mlp) {
    const float* w1p = (mlp == 0) ? W1 : sW1 + (long)(mlp-1) * DIN * HID;
    const float* b1p = (mlp == 0) ? b1 : sb1 + (mlp-1) * HID;
    const float* w2p = (mlp == 0) ? W2 : sW2 + (mlp-1) * HID * 3;
    const float* b2p = (mlp == 0) ? b2 : sb2 + (mlp-1) * 3;

    __syncthreads();   // previous iteration's LDS consumers are done

    // stage W1^T (f16, zero-padded K 39..63) + bias/W2 vectors into LDS
    for (int e = tid; e < DIN * HID; e += PTS) {
      const int k = e / HID, n = e % HID;      // coalesced read, n-major LDS write
      WT[n][k] = (f16)w1p[e];
    }
    for (int e = tid; e < (64 - DIN) * HID; e += PTS) {
      const int k = DIN + e / HID, n = e % HID;
      WT[n][k] = (f16)0.0f;
    }
    for (int e = tid; e < HID;     e += PTS) b1s[e] = b1p[e];
    for (int e = tid; e < HID * 3; e += PTS) w2s[e] = w2p[e];
    if (tid < 3) b2s[tid] = b2p[tid];

    // per-point interpolated feature row (L2-resident gathers)
    const unsigned mb = (mlp == 0) ? 0xffffffffu : (mlp == 1 ? mb0 : mb1);
    const float wscale = (mlp == 0) ? inv : (mlp == 1 ? rcp0 : rcp1);
    const float* fsrc  = (mlp == 0) ? cfeat : ecf;

    float facc[DF];
    #pragma unroll
    for (int d = 0; d < DF; ++d) facc[d] = 0.f;
    #pragma unroll
    for (int k = 0; k < KN; ++k) {
      if (mb & (1u << k)) {
        const float w = ev[k];
        const f4* r = (const f4*)(fsrc + (long)irow[k] * DF);
        #pragma unroll
        for (int q = 0; q < DF/4; ++q) {
          const f4 c = r[q];
          facc[q*4+0] += w * c.x;  facc[q*4+1] += w * c.y;
          facc[q*4+2] += w * c.z;  facc[q*4+3] += w * c.w;
        }
      }
    }

    float d0,d1,d2, q0,q1,q2, dmv;
    if (mlp == 0) { d0=vx; d1=vy; d2=vz; q0=nx; q1=ny; q2=nz; dmv=dsmean; }
    else {
      const float* R = rot + (mlp-1) * 9;
      d0 = R[0]*vx + R[1]*vy + R[2]*vz;
      d1 = R[3]*vx + R[4]*vy + R[5]*vz;
      d2 = R[6]*vx + R[7]*vy + R[8]*vz;
      q0 = R[0]*nx + R[1]*ny + R[2]*nz;
      q1 = R[3]*nx + R[4]*ny + R[5]*nz;
      q2 = R[6]*nx + R[7]*ny + R[8]*nz;
      dmv = ((mlp == 1) ? md0 : md1) * wscale;
    }
    f16* arow = &Atile[tid][0];
    #pragma unroll
    for (int d = 0; d < DF; ++d) arow[d] = (f16)(facc[d] * wscale);
    arow[32]=(f16)d0; arow[33]=(f16)d1; arow[34]=(f16)d2;
    arow[35]=(f16)q0; arow[36]=(f16)q1; arow[37]=(f16)q2;
    arow[38]=(f16)dmv;
    #pragma unroll
    for (int d = DIN; d < 64; ++d) arow[d] = (f16)0.0f;

    __syncthreads();

    // ---- WMMA: A(128x64) @ W1T -> H(128x256), fused relu+W2 epilogue ------
    const int khA = (lane >> 4) * 8;    // A fragment: K pairs 0-7 / 8-15 per half
    const int khB = (lane >> 4) * 8;    // B fragment: K pairs 0-7 / 8-15 per half
    const int nlane = lane & 15;
    for (int tt = 0; tt < 2; ++tt) {
      const int m0 = (wave * 2 + tt) * 16;
      const int ar = m0 + nlane;
      HV a0, a1;
      a0.h[0] = *(const h8*)&Atile[ar][ 0 + khA];
      a0.h[1] = *(const h8*)&Atile[ar][16 + khA];
      a1.h[0] = *(const h8*)&Atile[ar][32 + khA];
      a1.h[1] = *(const h8*)&Atile[ar][48 + khA];

      float racc[8][3];
      #pragma unroll
      for (int r = 0; r < 8; ++r) { racc[r][0]=0.f; racc[r][1]=0.f; racc[r][2]=0.f; }

      for (int n0 = 0; n0 < HID; n0 += 16) {
        const int n = n0 + nlane;
        HV b0, b1f;
        b0.h[0]  = *(const h8*)&WT[n][ 0 + khB];
        b0.h[1]  = *(const h8*)&WT[n][16 + khB];
        b1f.h[0] = *(const h8*)&WT[n][32 + khB];
        b1f.h[1] = *(const h8*)&WT[n][48 + khB];
        v8f c = {0.f,0.f,0.f,0.f,0.f,0.f,0.f,0.f};
        c = __builtin_amdgcn_wmma_f32_16x16x32_f16(false, a0.v, false, b0.v,
                                                   (short)0, c, false, false);
        c = __builtin_amdgcn_wmma_f32_16x16x32_f16(false, a1.v, false, b1f.v,
                                                   (short)0, c, false, false);
        const float bias = b1s[n];
        const float w20 = w2s[n*3+0], w21 = w2s[n*3+1], w22 = w2s[n*3+2];
        #pragma unroll
        for (int r = 0; r < 8; ++r) {
          float h = c[r] + bias;
          h = fmaxf(h, 0.f);
          racc[r][0] += h * w20;
          racc[r][1] += h * w21;
          racc[r][2] += h * w22;
        }
      }
      // reduce over the 16 N-lanes (xor < 16 keeps half-waves separate)
      #pragma unroll
      for (int off = 1; off < 16; off <<= 1) {
        #pragma unroll
        for (int r = 0; r < 8; ++r) {
          racc[r][0] += __shfl_xor(racc[r][0], off, 32);
          racc[r][1] += __shfl_xor(racc[r][1], off, 32);
          racc[r][2] += __shfl_xor(racc[r][2], off, 32);
        }
      }
      if ((lane & 15) == 0) {           // lane 0 -> rows m0..m0+7, lane 16 -> +8
        const int pbase = m0 + (lane >> 4) * 8;
        #pragma unroll
        for (int r = 0; r < 8; ++r) {
          const int pt = pbase + r;
          const bool wr = (mlp == 0) || (paintf[mlp-1][pt] != 0);
          if (wr) {
            rgbbuf[pt][0] = sigmoidf(racc[r][0] + b2s[0]);
            rgbbuf[pt][1] = sigmoidf(racc[r][1] + b2s[1]);
            rgbbuf[pt][2] = sigmoidf(racc[r][2] + b2s[2]);
          }
        }
      }
    }
  }

  __syncthreads();
  out_rgb[p*3+0] = rgbbuf[tid][0];
  out_rgb[p*3+1] = rgbbuf[tid][1];
  out_rgb[p*3+2] = rgbbuf[tid][2];
}

extern "C" void kernel_launch(void* const* d_in, const int* in_sizes, int n_in,
                              void* d_out, int out_size, void* d_ws, size_t ws_size,
                              hipStream_t stream) {
  const float* xyz  = (const float*)d_in[0];
  const float* vd   = (const float*)d_in[1];
  const float* mv   = (const float*)d_in[2];
  const float* geo  = (const float*)d_in[3];
  const float* cf   = (const float*)d_in[4];
  const float* ecf  = (const float*)d_in[5];
  const float* rot  = (const float*)d_in[6];
  const float* W1   = (const float*)d_in[7];
  const float* b1   = (const float*)d_in[8];
  const float* W2   = (const float*)d_in[9];
  const float* b2   = (const float*)d_in[10];
  const float* sW1  = (const float*)d_in[11];
  const float* sb1  = (const float*)d_in[12];
  const float* sW2  = (const float*)d_in[13];
  const float* sb2  = (const float*)d_in[14];
  const int*   idx  = (const int*)d_in[15];
  const int*   msk  = (const int*)d_in[16];

  const int N = in_sizes[0] / 3;   // 262144
  const int V = in_sizes[2] / 3;   // 100000

  float* out_sdf = (float*)d_out;          // sdf first, then (N,3) colors
  float* out_rgb = out_sdf + N;

  neumesh_fused<<<N / PTS, PTS, 0, stream>>>(
      xyz, vd, mv, geo, cf, ecf, rot,
      W1, b1, W2, b2, sW1, sb1, sW2, sb2,
      idx, msk, out_sdf, out_rgb, V);
}
